// SparseMoE_80376017977982
// MI455X (gfx1250) — matrix operand order repaired and checked by reading, not compiled, not run on hardware
//
#include <hip/hip_runtime.h>
#include <hip/hip_bf16.h>
#include <math.h>

#define E_EXPERTS 8
#define TOPK 2

typedef __attribute__((ext_vector_type(16))) __bf16 v16bf;
typedef __attribute__((ext_vector_type(8)))  float  v8f;

union FragBF {
  v16bf v;
  uint4 q[2];
};

// 16-bit A-matrix 16x32 fragment (row-major source, leading dim ld):
// lanes 0-15 : row = lane,    K chunks [0..7] (VGPR0-3) and [16..23] (VGPR4-7)
// lanes 16-31: row = lane-16, K chunks [8..15]           and [24..31]
__device__ __forceinline__ v16bf load_a_tile(const __bf16* tile, int ld) {
  const int lane = threadIdx.x & 31;
  const int r = lane & 15, hf = lane >> 4;
  const __bf16* p = tile + (size_t)r * ld + hf * 8;
  FragBF f;
  f.q[0] = *(const uint4*)(p);
  f.q[1] = *(const uint4*)(p + 16);
  return f.v;
}

// 16-bit B-matrix 32x16 fragment. B is stored K-contiguous (B^T row-major,
// leading dim ld = K): lanes 0-15 hold col=lane K=[0..7],[16..23], lanes 16-31 K=[8..15],[24..31].
__device__ __forceinline__ v16bf load_b_tile(const __bf16* tile, int ld) {
  const int lane = threadIdx.x & 31;
  const int r = lane & 15, hf = lane >> 4;
  const __bf16* p = tile + (size_t)r * ld + hf * 8;
  FragBF f;
  f.q[0] = *(const uint4*)(p);
  f.q[1] = *(const uint4*)(p + 16);
  return f.v;
}

// -------------------- router: softmax gates + top-2 --------------------
__global__ __launch_bounds__(128) void router_kernel(
    const float* __restrict__ x, const float* __restrict__ Wg,
    float* __restrict__ topw, int* __restrict__ topi, int T, int D) {
  const int t = blockIdx.x * 4 + (threadIdx.x >> 5);
  const int lane = threadIdx.x & 31;
  if (t >= T) return;
  const float* xt = x + (size_t)t * D;
  float acc[E_EXPERTS] = {};
  for (int c = lane; c < D; c += 32) {
    const float xv = xt[c];
#pragma unroll
    for (int e = 0; e < E_EXPERTS; ++e) acc[e] += xv * Wg[(size_t)e * D + c];
  }
#pragma unroll
  for (int e = 0; e < E_EXPERTS; ++e) {
#pragma unroll
    for (int off = 16; off > 0; off >>= 1) acc[e] += __shfl_xor(acc[e], off, 32);
  }
  if (lane == 0) {
    float mx = acc[0];
#pragma unroll
    for (int e = 1; e < E_EXPERTS; ++e) mx = fmaxf(mx, acc[e]);
    float g[E_EXPERTS], s = 0.f;
#pragma unroll
    for (int e = 0; e < E_EXPERTS; ++e) { g[e] = __expf(acc[e] - mx); s += g[e]; }
    const float inv = 1.0f / s;
#pragma unroll
    for (int e = 0; e < E_EXPERTS; ++e) g[e] *= inv;
    int i0 = 0;
#pragma unroll
    for (int e = 1; e < E_EXPERTS; ++e) if (g[e] > g[i0]) i0 = e;
    int i1 = (i0 == 0) ? 1 : 0;
#pragma unroll
    for (int e = 0; e < E_EXPERTS; ++e) if (e != i0 && g[e] > g[i1]) i1 = e;
    topi[t * TOPK + 0] = i0; topi[t * TOPK + 1] = i1;
    topw[t * TOPK + 0] = g[i0]; topw[t * TOPK + 1] = g[i1];
  }
}

// --------- slot-major capacity scan: exact reference arrival order ---------
// Single wave32; 2*T divisible by 32 (T = 4096).
__global__ void dispatch_pos_kernel(const int* __restrict__ topi,
                                    int* __restrict__ pos, int T, int C) {
  const int lane = threadIdx.x;
  const unsigned long long below = (1ull << lane) - 1ull;
  int cnt[E_EXPERTS] = {};
  const int A = TOPK * T;
  for (int a0 = 0; a0 < A; a0 += 32) {
    const int a = a0 + lane;
    const int k = a / T, t = a - k * T;
    const int e = topi[t * TOPK + k];
    int p = -1;
#pragma unroll
    for (int j = 0; j < E_EXPERTS; ++j) {
      const unsigned long long m = __ballot(e == j) & 0xffffffffull;
      if (e == j) {
        const int c = cnt[j] + __popcll(m & below);
        p = (c < C) ? c : -1;
      }
      cnt[j] += __popcll(m);
    }
    pos[a] = p;
  }
}

// -------------------- scatter accepted tokens into bf16 buffers --------------------
__global__ __launch_bounds__(256) void scatter_kernel(
    const float* __restrict__ x, const int* __restrict__ topi,
    const int* __restrict__ pos, __bf16* __restrict__ buf,
    int T, int D, int Mp) {
  const int a = blockIdx.x;
  const int p = pos[a];
  if (p < 0) return;
  const int k = a / T, t = a - k * T;
  const int e = topi[t * TOPK + k];
  const float* src = x + (size_t)t * D;
  __bf16* dst = buf + ((size_t)e * Mp + p) * D;
  for (int d = threadIdx.x; d < D; d += 256) dst[d] = (__bf16)src[d];
}

// ---------- f32 [R][C] -> bf16 [C][R] per-expert transpose+convert ----------
__global__ __launch_bounds__(256) void transpose_cvt_kernel(
    const float* __restrict__ in, __bf16* __restrict__ out, int R, int Cc) {
  __shared__ float tile[32][33];
  const float* ine = in + (size_t)blockIdx.z * R * Cc;
  __bf16* oute = out + (size_t)blockIdx.z * R * Cc;
  const int c0 = blockIdx.x * 32, r0 = blockIdx.y * 32;
  const int tx = threadIdx.x, ty = threadIdx.y; // 32 x 8
#pragma unroll
  for (int i = 0; i < 32; i += 8)
    tile[ty + i][tx] = ine[(size_t)(r0 + ty + i) * Cc + c0 + tx];
  __syncthreads();
#pragma unroll
  for (int i = 0; i < 32; i += 8)
    oute[(size_t)(c0 + ty + i) * R + r0 + tx] = (__bf16)tile[tx][ty + i];
}

// -------------------- grouped GEMM via v_wmma_f32_16x16x32_bf16 --------------------
// A: [E][Mp][K] bf16 row-major; Bt: [E][N][K] bf16 (K-contiguous); Out: [E][Mp][N]
// Block = 256 threads = 8 waves; block tile 128(M) x 128(N).
// Wave grid 4(M) x 2(N); wave tile 32 x 64 = 8 WMMA accumulators.
template <bool GELU, typename OutT>
__global__ __launch_bounds__(256) void moe_gemm_kernel(
    const __bf16* __restrict__ A, const __bf16* __restrict__ Bt,
    const float* __restrict__ bias, OutT* __restrict__ Out,
    int M, int N, int Kd) {
  __shared__ OutT smem[32 * 132]; // one 32-row stripe, padded vs bank conflicts
  const int e = blockIdx.z;
  const int m0 = blockIdx.y * 128;
  const int nBlk = blockIdx.x * 128;
  const int wave = threadIdx.x >> 5;
  const int lane = threadIdx.x & 31;
  const int wm = wave & 3;   // 0..3 -> M offset wm*32
  const int wn = wave >> 2;  // 0..1 -> N offset wn*64

  const __bf16* Am = A + ((size_t)e * M + m0 + wm * 32) * Kd;
  const __bf16* Bn = Bt + ((size_t)e * N + nBlk + wn * 64) * Kd;

  v8f acc[2][4] = {};
  for (int kb = 0; kb < Kd; kb += 32) {
    if (kb + 32 < Kd) {
      __builtin_prefetch(Am + kb + 32, 0, 1);
      __builtin_prefetch(Am + 16 * (size_t)Kd + kb + 32, 0, 1);
    }
    const v16bf a0 = load_a_tile(Am + kb, Kd);
    const v16bf a1 = load_a_tile(Am + 16 * (size_t)Kd + kb, Kd);
    v16bf bf[4];
#pragma unroll
    for (int j = 0; j < 4; ++j)
      bf[j] = load_b_tile(Bn + 16 * (size_t)j * Kd + kb, Kd);
#pragma unroll
    for (int j = 0; j < 4; ++j) {
      acc[0][j] = __builtin_amdgcn_wmma_f32_16x16x32_bf16(
          false, a0, false, bf[j], (short)0, acc[0][j], false, false);
      acc[1][j] = __builtin_amdgcn_wmma_f32_16x16x32_bf16(
          false, a1, false, bf[j], (short)0, acc[1][j], false, false);
    }
  }

  // epilogue: bias (+ exact GELU), staged through LDS one 32-row stripe at a time
  const int r = lane & 15, hf = lane >> 4;
#pragma unroll
  for (int p = 0; p < 4; ++p) {
    if (wm == p) {
#pragma unroll
      for (int mi = 0; mi < 2; ++mi) {
#pragma unroll
        for (int j = 0; j < 4; ++j) {
          const int col = wn * 64 + 16 * j + r;
          const float bv = bias[(size_t)e * N + nBlk + col];
#pragma unroll
          for (int i = 0; i < 8; ++i) {
            const int row = mi * 16 + 8 * hf + i;
            float v = acc[mi][j][i] + bv;
            if (GELU) v = 0.5f * v * (1.0f + erff(v * 0.70710678118654752f));
            smem[row * 132 + col] = (OutT)v;
          }
        }
      }
    }
    __syncthreads();
    // coalesced store of the 32x128 stripe: 4096 elems / 256 threads
#pragma unroll
    for (int it = 0; it < 16; ++it) {
      const int idx = it * 256 + (int)threadIdx.x;
      const int row = idx >> 7, col = idx & 127;
      Out[((size_t)e * M + m0 + p * 32 + row) * N + nBlk + col] =
          smem[row * 132 + col];
    }
    __syncthreads();
  }
}

// -------------------- weighted combine back to tokens --------------------
__global__ __launch_bounds__(256) void combine_kernel(
    const float* __restrict__ y, const int* __restrict__ topi,
    const float* __restrict__ topw, const int* __restrict__ pos,
    float* __restrict__ out, int T, int D, int Mp) {
  const int t = blockIdx.x;
  float w[TOPK];
  const float* src[TOPK];
#pragma unroll
  for (int k = 0; k < TOPK; ++k) {
    const int a = k * T + t;
    const int p = pos[a];
    if (p >= 0) {
      const int e = topi[t * TOPK + k];
      w[k] = topw[t * TOPK + k];
      src[k] = y + ((size_t)e * Mp + p) * D;
    } else {
      w[k] = 0.f;
      src[k] = y; // safe dummy; weight is zero
    }
  }
  for (int d = threadIdx.x; d < D; d += 256)
    out[(size_t)t * D + d] = w[0] * src[0][d] + w[1] * src[1][d];
}

static inline size_t align256(size_t x) { return (x + 255) & ~(size_t)255; }

extern "C" void kernel_launch(void* const* d_in, const int* in_sizes, int n_in,
                              void* d_out, int out_size, void* d_ws, size_t ws_size,
                              hipStream_t stream) {
  const float* x  = (const float*)d_in[0];
  const float* Wg = (const float*)d_in[1];
  const float* W1 = (const float*)d_in[2];
  const float* b1 = (const float*)d_in[3];
  const float* W2 = (const float*)d_in[4];
  const float* b2 = (const float*)d_in[5];
  float* out = (float*)d_out;

  const int D = in_sizes[1] / E_EXPERTS;          // 1024
  const int T = in_sizes[0] / D;                  // 4096
  const int H = in_sizes[3] / E_EXPERTS;          // 4096
  const int C = (int)((double)TOPK * T * 1.05 / E_EXPERTS + 0.5); // 1075
  const int Mp = (C + 127) & ~127;                // 1152 (9 x 128 block rows)

  // workspace carve-up
  uintptr_t base = (uintptr_t)d_ws;
  size_t off = 0;
  float* topw = (float*)(base + off); off = align256(off + (size_t)T * TOPK * 4);
  int*   topi = (int*)(base + off);   off = align256(off + (size_t)T * TOPK * 4);
  int*   pos  = (int*)(base + off);   off = align256(off + (size_t)T * TOPK * 4);
  __bf16* buf = (__bf16*)(base + off);
  const size_t bufBytes = (size_t)E_EXPERTS * Mp * D * 2;
  off = align256(off + bufBytes);
  __bf16* hbuf = (__bf16*)(base + off); off = align256(off + (size_t)E_EXPERTS * Mp * H * 2);
  float*  ybuf = (float*)(base + off);  off = align256(off + (size_t)E_EXPERTS * Mp * D * 4);
  __bf16* W1b = (__bf16*)(base + off);  off = align256(off + (size_t)E_EXPERTS * D * H * 2);
  __bf16* W2b = (__bf16*)(base + off);  off = align256(off + (size_t)E_EXPERTS * D * H * 2);
  (void)ws_size; (void)n_in; (void)out_size;

  // 1) router
  router_kernel<<<(T + 3) / 4, 128, 0, stream>>>(x, Wg, topw, topi, T, D);
  // 2) capacity scan (exact slot-major order)
  dispatch_pos_kernel<<<1, 32, 0, stream>>>(topi, pos, T, C);
  // 3) zero dispatch buffers, scatter tokens (f32 -> bf16)
  hipMemsetAsync(buf, 0, bufBytes, stream);
  scatter_kernel<<<T * TOPK, 256, 0, stream>>>(x, topi, pos, buf, T, D, Mp);
  // 4) weights -> bf16, transposed to K-contiguous for WMMA B fragments
  transpose_cvt_kernel<<<dim3(H / 32, D / 32, E_EXPERTS), dim3(32, 8), 0, stream>>>(W1, W1b, D, H);
  transpose_cvt_kernel<<<dim3(D / 32, H / 32, E_EXPERTS), dim3(32, 8), 0, stream>>>(W2, W2b, H, D);
  // 5) h = gelu(buf @ W1 + b1)   [E][Mp][H] bf16
  moe_gemm_kernel<true, __bf16>
      <<<dim3(H / 128, Mp / 128, E_EXPERTS), 256, 0, stream>>>(buf, W1b, b1, hbuf, Mp, H, D);
  // 6) y = h @ W2 + b2           [E][Mp][D] f32
  moe_gemm_kernel<false, float>
      <<<dim3(D / 128, Mp / 128, E_EXPERTS), 256, 0, stream>>>(hbuf, W2b, b2, ybuf, Mp, D, H);
  // 7) weighted combine
  combine_kernel<<<T, 256, 0, stream>>>(ybuf, topi, topw, pos, out, T, D, Mp);
}
